// MultiHeadedAttention_13692355740426
// MI455X (gfx1250) — matrix operand order repaired and checked by reading, not compiled, not run on hardware
//
#include <hip/hip_runtime.h>
#include <hip/hip_bf16.h>

typedef __attribute__((ext_vector_type(16))) _Float16 v16h;
typedef __attribute__((ext_vector_type(8)))  _Float16 v8h;
typedef __attribute__((ext_vector_type(8)))  float    v8f;

#define NTOK 2048
#define DMODEL 256
#define NHEAD 4
#define HDIM 64

// ---------------------------------------------------------------------------
// Projection GEMM:  Out = (W @ X + bias) * scale, converted to f16.
//   X: f32 [B][256][2048] (one batch per blockIdx.z)
//   W: f32 [256][256], bias f32 [256]
//   transpose==1: store Out[(b*4+h)*2048 + n][dd]   (token-major, dd contiguous)
//   transpose==0: store Out[(b*4+h)*64 + dd][n]     (dd-major, token contiguous)
// Workgroup: 256 threads (8 waves), tile M=64 (out channels) x 128 columns.
// Wave w: rows (w&3)*16, columns (w>>2)*64 -> 4 WMMA 16x16 tiles, K loop 256/32.
// ---------------------------------------------------------------------------
__global__ __launch_bounds__(256)
void proj_kernel(const float* __restrict__ X, const float* __restrict__ W,
                 const float* __restrict__ bias, _Float16* __restrict__ Out,
                 float scale, int transpose)
{
    const int b     = blockIdx.z;
    const int n0    = blockIdx.x * 128;
    const int obase = blockIdx.y * 64;
    const int tid   = threadIdx.x;
    const int lane  = tid & 31;
    const int w     = tid >> 5;
    const int grow  = w & 3;       // 16-row group inside 64-row tile
    const int chalf = w >> 2;      // which 64-column half
    const int lcol  = lane & 15;
    const int lh    = lane >> 4;   // lane half

    __shared__ _Float16 Xs[128 * 48];   // [n local 0..127][c local 0..31], stride 48

    v8f acc[4];
    acc[0] = {}; acc[1] = {}; acc[2] = {}; acc[3] = {};

    for (int kk = 0; kk < DMODEL; kk += 32) {
        // ---- stage X tile [32 c][128 n] into LDS transposed as f16 [n][c] ----
        {
            const int c     = tid >> 3;   // 0..31
            const int chunk = tid & 7;    // 16 n's each
            const float* xp = X + (((size_t)b * DMODEL + kk + c) * NTOK + n0 + chunk * 16);
            #pragma unroll
            for (int i4 = 0; i4 < 4; ++i4) {
                float4 v = ((const float4*)xp)[i4];
                const int nl = chunk * 16 + i4 * 4;
                Xs[(nl + 0) * 48 + c] = (_Float16)v.x;
                Xs[(nl + 1) * 48 + c] = (_Float16)v.y;
                Xs[(nl + 2) * 48 + c] = (_Float16)v.z;
                Xs[(nl + 3) * 48 + c] = (_Float16)v.w;
            }
        }
        __syncthreads();

        // ---- A fragment from W (f32 -> f16): row o, K = c ----
        v16h a;
        {
            const int o = obase + grow * 16 + lcol;
            const float* wp = W + ((size_t)o * DMODEL + kk);
            #pragma unroll
            for (int i = 0; i < 8; ++i) {
                a[i]     = (_Float16)wp[lh * 16 + i];
                a[8 + i] = (_Float16)wp[lh * 16 + 8 + i];
            }
        }

        // ---- 4 column tiles ----
        #pragma unroll
        for (int t = 0; t < 4; ++t) {
            const int nl = chalf * 64 + t * 16 + lcol;
            v16h bf = *(const v16h*)&Xs[nl * 48 + lh * 16];  // K = lh*16 + e, contiguous
            acc[t] = __builtin_amdgcn_wmma_f32_16x16x32_f16(
                false, a, false, bf, (short)0, acc[t], false, false);
        }
        __syncthreads();
    }

    // ---- epilogue: bias, scale, f16 store in consumer layout ----
    #pragma unroll
    for (int t = 0; t < 4; ++t) {
        #pragma unroll
        for (int r = 0; r < 8; ++r) {
            const int o  = obase + grow * 16 + lh * 8 + r;
            const int n  = n0 + chalf * 64 + t * 16 + lcol;
            const int dd = o >> 2, h = o & 3;
            const float val = (acc[t][r] + bias[o]) * scale;
            const size_t idx = transpose
                ? (((size_t)(b * NHEAD + h) * NTOK + n) * HDIM + dd)
                : (((size_t)(b * NHEAD + h) * HDIM + dd) * NTOK + n);
            Out[idx] = (_Float16)val;
        }
    }
}

// ---------------------------------------------------------------------------
// Flash attention per (b,h). Query block of 64 rows per workgroup.
//   Qt,Kt: f16 [bh][2048][64]  (token-major, dd contiguous; Q pre-scaled 1/8)
//   Vh:    f16 [bh][64][2048]  (dd-major, token contiguous)
//   Xa:    f16 [b][2048][256]  (c = dd*4 + h)
// 8 waves: row group g = w&3 (16 query rows), half hh = w>>2
//   S:  wave covers [16 n][32 m = hh half], O: [16 n][32 dd = hh half]
// Online softmax is parallelized 4 threads/row (all 256 threads active).
// ---------------------------------------------------------------------------
__global__ __launch_bounds__(256)
void attn_kernel(const _Float16* __restrict__ Qt, const _Float16* __restrict__ Kt,
                 const _Float16* __restrict__ Vh, const int* __restrict__ mask,
                 _Float16* __restrict__ Xa)
{
    const int nb   = blockIdx.x;          // 64-row query block
    const int bh   = blockIdx.y;          // b*4 + h
    const int b    = bh >> 2, h = bh & 3;
    const int tid  = threadIdx.x;
    const int lane = tid & 31, w = tid >> 5;
    const int g    = w & 3;
    const int hh   = w >> 2;
    const int lcol = lane & 15, lh = lane >> 4;

    __shared__ float    Ss[64 * 68];      // raw scores (f32)
    __shared__ _Float16 Ps[64 * 72];      // softmax numerator (f16), 16B-aligned rows
    __shared__ float    rowmaxS[64], rowsumS[64], alphaS[64];
    __shared__ float    pmaxS[64 * 4], psumS[64 * 4];

    const size_t qkbase = (size_t)bh * NTOK * HDIM;
    const size_t vbase  = (size_t)bh * HDIM * NTOK;

    // Q A-fragments for both K-steps (dd 0..31, 32..63), resident all kernel
    v16h aq[2];
    {
        const int n = nb * 64 + g * 16 + lcol;
        const _Float16* qp = Qt + qkbase + (size_t)n * HDIM;
        #pragma unroll
        for (int ks = 0; ks < 2; ++ks) {
            v8h lo = *(const v8h*)(qp + ks * 32 + lh * 16);
            v8h hi = *(const v8h*)(qp + ks * 32 + lh * 16 + 8);
            #pragma unroll
            for (int i = 0; i < 8; ++i) { aq[ks][i] = lo[i]; aq[ks][8 + i] = hi[i]; }
        }
    }

    v8f oacc[2];
    oacc[0] = {}; oacc[1] = {};

    if (tid < 64) { rowmaxS[tid] = -1e30f; rowsumS[tid] = 0.f; }
    __syncthreads();

    for (int mb = 0; mb < NTOK / 64; ++mb) {
        const int m0 = mb * 64;

        // ---- prefetch next m-block K/V rows (global_prefetch_b8) ----
        if (mb + 1 < NTOK / 64) {
            if (tid < 64) {
                const int mn = m0 + 64 + tid;
                __builtin_prefetch(Kt + qkbase + (size_t)mn * HDIM, 0, 3);
            } else if (tid < 128) {
                __builtin_prefetch(Vh + vbase + (size_t)(tid - 64) * NTOK + m0 + 64, 0, 3);
            }
        }

        // ---- S = (Q/8)^T K for this wave's [16 n][32 m] slice ----
        #pragma unroll
        for (int t = 0; t < 2; ++t) {
            v8f sacc = {};
            const int m = m0 + hh * 32 + t * 16 + lcol;
            const _Float16* kp = Kt + qkbase + (size_t)m * HDIM;
            #pragma unroll
            for (int ks = 0; ks < 2; ++ks) {
                v16h bk = *(const v16h*)(kp + ks * 32 + lh * 16);
                sacc = __builtin_amdgcn_wmma_f32_16x16x32_f16(
                    false, aq[ks], false, bk, (short)0, sacc, false, false);
            }
            #pragma unroll
            for (int r = 0; r < 8; ++r)
                Ss[(g * 16 + lh * 8 + r) * 68 + hh * 32 + t * 16 + lcol] = sacc[r];
        }
        __syncthreads();

        // ---- online softmax: 4 threads per row, scores held in registers ----
        {
            const int row  = tid >> 2;     // 0..63
            const int part = tid & 3;      // which 16-column slice
            const int c0   = part * 16;
            const int nglob = nb * 64 + row;
            const int mq = mask[b * NTOK + nglob];

            float sv[16];
            float lmax = -1e30f;
            #pragma unroll
            for (int i = 0; i < 16; ++i) {
                float s = Ss[row * 68 + c0 + i];
                if (mq == 0 || mask[b * NTOK + m0 + c0 + i] == 0) s = -1e9f;
                sv[i] = s;
                lmax = fmaxf(lmax, s);
            }
            pmaxS[row * 4 + part] = lmax;
            __syncthreads();

            const float bm = fmaxf(fmaxf(pmaxS[row * 4 + 0], pmaxS[row * 4 + 1]),
                                   fmaxf(pmaxS[row * 4 + 2], pmaxS[row * 4 + 3]));
            const float nm = fmaxf(rowmaxS[row], bm);

            float lsum = 0.f;
            #pragma unroll
            for (int i = 0; i < 16; ++i) {
                float p = __expf(sv[i] - nm);
                lsum += p;
                Ps[row * 72 + c0 + i] = (_Float16)p;
            }
            psumS[row * 4 + part] = lsum;
            __syncthreads();

            if (part == 0) {
                const float om = rowmaxS[row];
                const float al = __expf(om - nm);
                const float sum = psumS[row * 4 + 0] + psumS[row * 4 + 1] +
                                  psumS[row * 4 + 2] + psumS[row * 4 + 3];
                rowmaxS[row] = nm;
                rowsumS[row] = rowsumS[row] * al + sum;
                alphaS[row]  = al;
            }
        }
        __syncthreads();

        // ---- rescale running O by alpha ----
        #pragma unroll
        for (int t = 0; t < 2; ++t)
            #pragma unroll
            for (int r = 0; r < 8; ++r)
                oacc[t][r] *= alphaS[g * 16 + lh * 8 + r];

        // ---- O += P @ V^T  (K dim = m, two 32-steps) ----
        v16h ap[2];
        {
            const int row = g * 16 + lcol;
            #pragma unroll
            for (int ks = 0; ks < 2; ++ks) {
                v8h lo = *(const v8h*)&Ps[row * 72 + ks * 32 + lh * 16];
                v8h hi = *(const v8h*)&Ps[row * 72 + ks * 32 + lh * 16 + 8];
                #pragma unroll
                for (int i = 0; i < 8; ++i) { ap[ks][i] = lo[i]; ap[ks][8 + i] = hi[i]; }
            }
        }
        #pragma unroll
        for (int t = 0; t < 2; ++t) {
            const int dd = hh * 32 + t * 16 + lcol;
            const _Float16* vp = Vh + vbase + (size_t)dd * NTOK + m0;
            #pragma unroll
            for (int ks = 0; ks < 2; ++ks) {
                v16h bv = *(const v16h*)(vp + ks * 32 + lh * 16);
                oacc[t] = __builtin_amdgcn_wmma_f32_16x16x32_f16(
                    false, ap[ks], false, bv, (short)0, oacc[t], false, false);
            }
        }
        __syncthreads();
    }

    // ---- normalize and store as [b][n][c], c = dd*4 + h ----
    #pragma unroll
    for (int t = 0; t < 2; ++t) {
        #pragma unroll
        for (int r = 0; r < 8; ++r) {
            const int row = g * 16 + lh * 8 + r;
            const int n   = nb * 64 + row;
            const int dd  = hh * 32 + t * 16 + lcol;
            const float val = oacc[t][r] / fmaxf(rowsumS[row], 1e-20f);
            Xa[((size_t)b * NTOK + n) * DMODEL + dd * NHEAD + h] = (_Float16)val;
        }
    }
}

// ---------------------------------------------------------------------------
// Final merge GEMM: Out[b][o][n] = sum_c Wm[o][c] * Xa[b][n][c] + bm[o]  (f32 out)
// Xa already has c contiguous -> B fragments are direct 32B global loads.
// ---------------------------------------------------------------------------
__global__ __launch_bounds__(256)
void out_kernel(const _Float16* __restrict__ Xa, const float* __restrict__ W,
                const float* __restrict__ bias, float* __restrict__ Out)
{
    const int b     = blockIdx.z;
    const int n0    = blockIdx.x * 128;
    const int obase = blockIdx.y * 64;
    const int tid   = threadIdx.x;
    const int lane  = tid & 31;
    const int w     = tid >> 5;
    const int grow  = w & 3;
    const int chalf = w >> 2;
    const int lcol  = lane & 15;
    const int lh    = lane >> 4;

    v8f acc[4];
    acc[0] = {}; acc[1] = {}; acc[2] = {}; acc[3] = {};

    for (int kk = 0; kk < DMODEL; kk += 32) {
        v16h a;
        {
            const int o = obase + grow * 16 + lcol;
            const float* wp = W + ((size_t)o * DMODEL + kk);
            #pragma unroll
            for (int i = 0; i < 8; ++i) {
                a[i]     = (_Float16)wp[lh * 16 + i];
                a[8 + i] = (_Float16)wp[lh * 16 + 8 + i];
            }
        }
        #pragma unroll
        for (int t = 0; t < 4; ++t) {
            const int n = n0 + chalf * 64 + t * 16 + lcol;
            v16h bf = *(const v16h*)(Xa + ((size_t)b * NTOK + n) * DMODEL + kk + lh * 16);
            acc[t] = __builtin_amdgcn_wmma_f32_16x16x32_f16(
                false, a, false, bf, (short)0, acc[t], false, false);
        }
    }

    #pragma unroll
    for (int t = 0; t < 4; ++t) {
        #pragma unroll
        for (int r = 0; r < 8; ++r) {
            const int o = obase + grow * 16 + lh * 8 + r;
            const int n = n0 + chalf * 64 + t * 16 + lcol;
            Out[((size_t)b * DMODEL + o) * NTOK + n] = acc[t][r] + bias[o];
        }
    }
}

// ---------------------------------------------------------------------------
extern "C" void kernel_launch(void* const* d_in, const int* in_sizes, int n_in,
                              void* d_out, int out_size, void* d_ws, size_t ws_size,
                              hipStream_t stream) {
    const float* query = (const float*)d_in[0];
    const float* key   = (const float*)d_in[1];
    const float* value = (const float*)d_in[2];
    const int*   mask  = (const int*)d_in[3];
    const float* Wq = (const float*)d_in[4];
    const float* bq = (const float*)d_in[5];
    const float* Wk = (const float*)d_in[6];
    const float* bk = (const float*)d_in[7];
    const float* Wv = (const float*)d_in[8];
    const float* bv = (const float*)d_in[9];
    const float* Wm = (const float*)d_in[10];
    const float* bm = (const float*)d_in[11];
    float* out = (float*)d_out;

    // workspace: 4 x 8MB f16 buffers (Qt, Kt, Vh, Xa) = 32 MB
    _Float16* ws = (_Float16*)d_ws;
    const size_t per = (size_t)8 * NHEAD * NTOK * HDIM;  // 4Mi elements
    _Float16* Qt = ws;
    _Float16* Kt = ws + per;
    _Float16* Vh = ws + 2 * per;
    _Float16* Xa = ws + 3 * per;

    dim3 blk(256);
    dim3 gproj(NTOK / 128, DMODEL / 64, 8);
    // Q gets the 1/sqrt(64) score scale folded in; Q,K token-major, V dd-major
    proj_kernel<<<gproj, blk, 0, stream>>>(query, Wq, bq, Qt, 0.125f, 1);
    proj_kernel<<<gproj, blk, 0, stream>>>(key,   Wk, bk, Kt, 1.0f,   1);
    proj_kernel<<<gproj, blk, 0, stream>>>(value, Wv, bv, Vh, 1.0f,   0);

    dim3 gattn(NTOK / 64, 8 * NHEAD);
    attn_kernel<<<gattn, blk, 0, stream>>>(Qt, Kt, Vh, mask, Xa);

    out_kernel<<<gproj, blk, 0, stream>>>(Xa, Wm, bm, out);
}
